// EnhancedHeteroGraph_6305011991143
// MI455X (gfx1250) — matrix operand order repaired and checked by reading, not compiled, not run on hardware
//
#include <hip/hip_runtime.h>
#include <hip/hip_bf16.h>
#include <math.h>
#include <stdint.h>

// ---------------------------------------------------------------------------
// Types for CDNA5 WMMA
// ---------------------------------------------------------------------------
typedef __attribute__((ext_vector_type(16))) __bf16 v16bf;
typedef __attribute__((ext_vector_type(8)))  float  v8f;

union FragB16 {
    v16bf v;
    unsigned int u[8];
    uint4 q[2];
};

__device__ inline void atomic_add_f32(float* p, float v) {
#if defined(__AMDGCN__)
    unsafeAtomicAdd(p, v);   // lowers to global_atomic_add_f32
#else
    atomicAdd(p, v);
#endif
}

// Split fp32 a = hi + lo, both representable as bf16 (truncation split; the
// subtraction a - hi is exact in fp32, so the 3-term bf16 product recovers
// ~2^-16 relative error).
__device__ inline void split_bf16(float a, unsigned short* ph, unsigned short* pl) {
    unsigned int u = __float_as_uint(a);
    *ph = (unsigned short)(u >> 16);
    float hi = __uint_as_float(u & 0xFFFF0000u);
    float lo = a - hi;
    *pl = (unsigned short)(__float_as_uint(lo) >> 16);
}

// ---------------------------------------------------------------------------
// gfx1250 async global->LDS copy (GLOBAL_LOAD_ASYNC_TO_LDS_B128, ASYNCcnt),
// with a plain-copy fallback that always compiles.
// Builtin signature (from hipcc diagnostics): param0 is
//   'int __vector(4) __device__ *'  i.e. addrspace(1) pointer to int4,
// param1 is the LDS (addrspace(3)) pointer, then imm offset + cpol.
// ---------------------------------------------------------------------------
#if __has_builtin(__builtin_amdgcn_global_load_async_to_lds_b128) && \
    __has_builtin(__builtin_amdgcn_s_wait_asynccnt)
#define USE_ASYNC_LDS 1
#else
#define USE_ASYNC_LDS 0
#endif

#if USE_ASYNC_LDS
typedef int v4i_gnu __attribute__((vector_size(4 * sizeof(int))));
typedef __attribute__((address_space(1))) v4i_gnu async_global_t;
typedef __attribute__((address_space(3))) v4i_gnu async_lds_t;
#endif

__device__ inline void cp16(void* lds, const void* g) {
#if USE_ASYNC_LDS
    __builtin_amdgcn_global_load_async_to_lds_b128(
        (async_global_t*)(unsigned long long)(uintptr_t)g,
        (async_lds_t*)(unsigned)(uintptr_t)lds,
        0, 0);
#else
    *(uint4*)lds = *(const uint4*)g;
#endif
}

__device__ inline void cp_wait_all() {
#if USE_ASYNC_LDS
    __builtin_amdgcn_s_wait_asynccnt(0);
#endif
}

// ---------------------------------------------------------------------------
// Pack kernels: fp32 -> split-bf16 planes (done once per call)
// ---------------------------------------------------------------------------
__global__ void pack_split_kernel(const float* __restrict__ src,
                                  unsigned short* __restrict__ hi,
                                  unsigned short* __restrict__ lo, long n) {
    long i = ((long)blockIdx.x * blockDim.x + threadIdx.x) * 4;
    if (i + 3 < n) {
        float4 v = *(const float4*)(src + i);
        unsigned short h[4], l[4];
        split_bf16(v.x, &h[0], &l[0]);
        split_bf16(v.y, &h[1], &l[1]);
        split_bf16(v.z, &h[2], &l[2]);
        split_bf16(v.w, &h[3], &l[3]);
        uint2 hp, lp;
        hp.x = (unsigned)h[0] | ((unsigned)h[1] << 16);
        hp.y = (unsigned)h[2] | ((unsigned)h[3] << 16);
        lp.x = (unsigned)l[0] | ((unsigned)l[1] << 16);
        lp.y = (unsigned)l[2] | ((unsigned)l[3] << 16);
        *(uint2*)(hi + i) = hp;
        *(uint2*)(lo + i) = lp;
    } else {
        for (long j = i; j < n; ++j) split_bf16(src[j], &hi[j], &lo[j]);
    }
}

// We [k][n] fp32 -> transposed split planes Wh/Wl [n][k]
__global__ void pack_w_kernel(const float* __restrict__ W,
                              unsigned short* __restrict__ Wh,
                              unsigned short* __restrict__ Wl, int D) {
    const int k = blockIdx.y;
    const int n = blockIdx.x * blockDim.x + threadIdx.x;
    if (n < D) {
        unsigned short h, l;
        split_bf16(W[(long)k * D + n], &h, &l);
        Wh[(long)n * D + k] = h;
        Wl[(long)n * D + k] = l;
    }
}

// ---------------------------------------------------------------------------
// Kernel A: per-graph mean of node features (node_graph sorted, 64 graphs)
// ---------------------------------------------------------------------------
__device__ inline int lower_bound_i(const int* a, int n, int v) {
    int lo = 0, hi = n;
    while (lo < hi) {
        int mid = (lo + hi) >> 1;
        if (a[mid] < v) lo = mid + 1; else hi = mid;
    }
    return lo;
}

__global__ void graph_mean_kernel(const float* __restrict__ nf,
                                  const int* __restrict__ ng,
                                  float* __restrict__ gmean,
                                  int N, int D) {
    const int b  = blockIdx.x;
    const int lo = lower_bound_i(ng, N, b);
    const int hi = lower_bound_i(ng, N, b + 1);
    const float inv = (hi > lo) ? 1.0f / (float)(hi - lo) : 0.0f;
    for (int c = threadIdx.x; c < D; c += blockDim.x) {
        float s = 0.0f;
        for (int r = lo; r < hi; ++r) s += nf[(long)r * D + c];
        gmean[(long)b * D + c] = s * inv;
    }
}

// ---------------------------------------------------------------------------
// Kernel B: virtual-node MLP (one block per graph, 512 threads)
// ---------------------------------------------------------------------------
__global__ __launch_bounds__(512) void vn_mlp_kernel(
    const float* __restrict__ gmean, const float* __restrict__ vn_emb,
    const float* __restrict__ W1, const float* __restrict__ b1,
    const float* __restrict__ ln_g, const float* __restrict__ ln_b,
    const float* __restrict__ W2, const float* __restrict__ b2,
    float* __restrict__ vn_out, int D) {
    __shared__ float sx[1024];
    __shared__ float shn[512];
    __shared__ float red[512];

    const int b = blockIdx.x;
    const int j = threadIdx.x;

    if (j < D) {
        sx[j]     = vn_emb[j];
        sx[D + j] = gmean[(long)b * D + j];
    }
    __syncthreads();

    float h = 0.0f;
    if (j < D) {
        h = b1[j];
        for (int k = 0; k < 2 * D; ++k) h += sx[k] * W1[(long)k * D + j];
        h = 0.5f * h * (1.0f + erff(h * 0.70710678118654752440f));  // exact GELU
    }

    red[j] = (j < D) ? h : 0.0f;
    __syncthreads();
    for (int off = 256; off > 0; off >>= 1) {
        if (j < off) red[j] += red[j + off];
        __syncthreads();
    }
    const float mu = red[0] / (float)D;
    __syncthreads();
    const float d = h - mu;
    red[j] = (j < D) ? d * d : 0.0f;
    __syncthreads();
    for (int off = 256; off > 0; off >>= 1) {
        if (j < off) red[j] += red[j + off];
        __syncthreads();
    }
    const float var = red[0] / (float)D;
    __syncthreads();

    if (j < D) shn[j] = d * rsqrtf(var + 1e-5f) * ln_g[j] + ln_b[j];
    __syncthreads();

    if (j < D) {
        float o = b2[j];
        for (int k = 0; k < D; ++k) o += shn[k] * W2[(long)k * D + j];
        vn_out[(long)b * D + j] = o;
    }
}

// ---------------------------------------------------------------------------
// Kernel C: edge in-degree counts
// ---------------------------------------------------------------------------
__global__ void edge_count_kernel(const int* __restrict__ dst,
                                  float* __restrict__ cnt, int E) {
    int t = blockIdx.x * blockDim.x + threadIdx.x;
    if (t < E) atomic_add_f32(&cnt[dst[t]], 1.0f);
}

// ---------------------------------------------------------------------------
// GEMM tiling
// ---------------------------------------------------------------------------
#define BM 128
#define BN 64
#define BK 32

// ---------------------------------------------------------------------------
// Kernel D (packed path): e = Apacked @ Wpacked^T + be, fused scatter-add.
// Double-buffered LDS, async global->LDS staging, 12 WMMAs per K-step.
// ---------------------------------------------------------------------------
__global__ __launch_bounds__(256) void edge_gemm_scatter_packed_kernel(
    const unsigned short* __restrict__ Ah, const unsigned short* __restrict__ Al,
    const unsigned short* __restrict__ Wh, const unsigned short* __restrict__ Wl,
    const float* __restrict__ be, const int* __restrict__ dst,
    float* __restrict__ e_sum, int E, int D) {
    __shared__ __align__(16) unsigned short sAh[2][BM][BK];
    __shared__ __align__(16) unsigned short sAl[2][BM][BK];
    __shared__ __align__(16) unsigned short sBh[2][BN][BK];
    __shared__ __align__(16) unsigned short sBl[2][BN][BK];

    const int tid  = threadIdx.x;
    const int lane = tid & 31;
    const int wave = tid >> 5;
    const int l15  = lane & 15;
    const int kh   = lane >> 4;
    const int n0   = blockIdx.x * BN;
    const long m0  = (long)blockIdx.y * BM;

    v8f acc[4];
#pragma unroll
    for (int s = 0; s < 4; ++s)
#pragma unroll
        for (int i = 0; i < 8; ++i) acc[s][i] = 0.0f;

    auto stage = [&](int kt, int p) {
        const int k0 = kt * BK;
#pragma unroll
        for (int r = 0; r < 2; ++r) {
            const int c   = tid + r * 256;      // 0..511
            const int row = c >> 2;             // 0..127
            const int q   = (c & 3) * 8;        // ushort offset, 16B chunks
            const long ge = m0 + row;
            if (ge < E) {
                cp16(&sAh[p][row][q], Ah + ge * D + k0 + q);
                cp16(&sAl[p][row][q], Al + ge * D + k0 + q);
            } else {
                const uint4 z = make_uint4(0u, 0u, 0u, 0u);
                *(uint4*)&sAh[p][row][q] = z;
                *(uint4*)&sAl[p][row][q] = z;
            }
        }
        {
            const int row = tid >> 2;           // 0..63
            const int q   = (tid & 3) * 8;
            cp16(&sBh[p][row][q], Wh + (size_t)(n0 + row) * D + k0 + q);
            cp16(&sBl[p][row][q], Wl + (size_t)(n0 + row) * D + k0 + q);
        }
    };

    const int kIters = D / BK;
    stage(0, 0);

    for (int kt = 0; kt < kIters; ++kt) {
        const int p = kt & 1;
        cp_wait_all();
        __syncthreads();
        if (kt + 1 < kIters) stage(kt + 1, p ^ 1);

        const int arow = wave * 16 + l15;
        FragB16 ah, al;
        ah.q[0] = *(const uint4*)&sAh[p][arow][8 * kh];
        ah.q[1] = *(const uint4*)&sAh[p][arow][16 + 8 * kh];
        al.q[0] = *(const uint4*)&sAl[p][arow][8 * kh];
        al.q[1] = *(const uint4*)&sAl[p][arow][16 + 8 * kh];

#pragma unroll
        for (int s = 0; s < 4; ++s) {
            const int bn = s * 16 + l15;
            FragB16 bh, bl;
            bh.q[0] = *(const uint4*)&sBh[p][bn][8 * kh];
            bh.q[1] = *(const uint4*)&sBh[p][bn][16 + 8 * kh];
            bl.q[0] = *(const uint4*)&sBl[p][bn][8 * kh];
            bl.q[1] = *(const uint4*)&sBl[p][bn][16 + 8 * kh];
            acc[s] = __builtin_amdgcn_wmma_f32_16x16x32_bf16(
                false, ah.v, false, bh.v, (short)0, acc[s], false, false);
            acc[s] = __builtin_amdgcn_wmma_f32_16x16x32_bf16(
                false, ah.v, false, bl.v, (short)0, acc[s], false, false);
            acc[s] = __builtin_amdgcn_wmma_f32_16x16x32_bf16(
                false, al.v, false, bh.v, (short)0, acc[s], false, false);
        }
    }

    // epilogue: C layout (VGPR r -> row, lane -> col); scatter by dst
    const int mbase = wave * 16 + (kh ? 8 : 0);
    int  dsts[8];
    bool valid[8];
#pragma unroll
    for (int r = 0; r < 8; ++r) {
        const long ge = m0 + mbase + r;
        valid[r] = (ge < (long)E);
        dsts[r]  = valid[r] ? dst[ge] : 0;
    }
#pragma unroll
    for (int s = 0; s < 4; ++s) {
        const int col    = n0 + s * 16 + l15;
        const float bias = be[col];
#pragma unroll
        for (int r = 0; r < 8; ++r) {
            if (valid[r])
                atomic_add_f32(&e_sum[(long)dsts[r] * D + col], acc[s][r] + bias);
        }
    }
}

// ---------------------------------------------------------------------------
// Kernel D (fallback path, fused fp32->bf16 conversion; used only if the
// workspace cannot hold the packed planes)
// ---------------------------------------------------------------------------
__global__ __launch_bounds__(256) void edge_gemm_scatter_fused_kernel(
    const float* __restrict__ Ef, const float* __restrict__ We,
    const float* __restrict__ be, const int* __restrict__ dst,
    float* __restrict__ e_sum, int E, int D) {
    __shared__ __align__(16) unsigned short sAh[BM][BK];
    __shared__ __align__(16) unsigned short sAl[BM][BK];
    __shared__ __align__(16) unsigned short sBh[BN][BK];
    __shared__ __align__(16) unsigned short sBl[BN][BK];

    const int tid  = threadIdx.x;
    const int lane = tid & 31;
    const int wave = tid >> 5;
    const int l15  = lane & 15;
    const int kh   = lane >> 4;
    const int n0   = blockIdx.x * BN;
    const long m0  = (long)blockIdx.y * BM;

    v8f acc[4];
#pragma unroll
    for (int s = 0; s < 4; ++s)
#pragma unroll
        for (int i = 0; i < 8; ++i) acc[s][i] = 0.0f;

    const int kIters = D / BK;
    for (int kt = 0; kt < kIters; ++kt) {
        const int k0 = kt * BK;
        __syncthreads();
#pragma unroll
        for (int r = 0; r < 4; ++r) {
            const int f4  = tid + r * 256;
            const int row = f4 >> 3;
            const int kq  = (f4 & 7) << 2;
            const long ge = m0 + row;
            float4 v = make_float4(0.f, 0.f, 0.f, 0.f);
            if (ge < E) v = *(const float4*)(Ef + ge * D + (k0 + kq));
            split_bf16(v.x, &sAh[row][kq + 0], &sAl[row][kq + 0]);
            split_bf16(v.y, &sAh[row][kq + 1], &sAl[row][kq + 1]);
            split_bf16(v.z, &sAh[row][kq + 2], &sAl[row][kq + 2]);
            split_bf16(v.w, &sAh[row][kq + 3], &sAl[row][kq + 3]);
        }
#pragma unroll
        for (int r = 0; r < 2; ++r) {
            const int f4 = tid + r * 256;
            const int kk = f4 >> 4;
            const int nn = (f4 & 15) << 2;
            const float4 v = *(const float4*)(We + (long)(k0 + kk) * D + n0 + nn);
            split_bf16(v.x, &sBh[nn + 0][kk], &sBl[nn + 0][kk]);
            split_bf16(v.y, &sBh[nn + 1][kk], &sBl[nn + 1][kk]);
            split_bf16(v.z, &sBh[nn + 2][kk], &sBl[nn + 2][kk]);
            split_bf16(v.w, &sBh[nn + 3][kk], &sBl[nn + 3][kk]);
        }
        __syncthreads();

        const int arow = wave * 16 + l15;
        FragB16 ah, al;
#pragma unroll
        for (int j = 0; j < 8; ++j) {
            const int kk = (j < 4) ? (8 * kh + 2 * j) : (16 + 8 * kh + 2 * (j - 4));
            ah.u[j] = *(const unsigned int*)&sAh[arow][kk];
            al.u[j] = *(const unsigned int*)&sAl[arow][kk];
        }
#pragma unroll
        for (int s = 0; s < 4; ++s) {
            const int bn = s * 16 + l15;
            FragB16 bh, bl;
#pragma unroll
            for (int j = 0; j < 8; ++j) {
                const int kk = (j < 4) ? (8 * kh + 2 * j) : (16 + 8 * kh + 2 * (j - 4));
                bh.u[j] = *(const unsigned int*)&sBh[bn][kk];
                bl.u[j] = *(const unsigned int*)&sBl[bn][kk];
            }
            acc[s] = __builtin_amdgcn_wmma_f32_16x16x32_bf16(
                false, ah.v, false, bh.v, (short)0, acc[s], false, false);
            acc[s] = __builtin_amdgcn_wmma_f32_16x16x32_bf16(
                false, ah.v, false, bl.v, (short)0, acc[s], false, false);
            acc[s] = __builtin_amdgcn_wmma_f32_16x16x32_bf16(
                false, al.v, false, bh.v, (short)0, acc[s], false, false);
        }
    }

    const int mbase = wave * 16 + (kh ? 8 : 0);
    int  dsts[8];
    bool valid[8];
#pragma unroll
    for (int r = 0; r < 8; ++r) {
        const long ge = m0 + mbase + r;
        valid[r] = (ge < (long)E);
        dsts[r]  = valid[r] ? dst[ge] : 0;
    }
#pragma unroll
    for (int s = 0; s < 4; ++s) {
        const int col    = n0 + s * 16 + l15;
        const float bias = be[col];
#pragma unroll
        for (int r = 0; r < 8; ++r) {
            if (valid[r])
                atomic_add_f32(&e_sum[(long)dsts[r] * D + col], acc[s][r] + bias);
        }
    }
}

// ---------------------------------------------------------------------------
// Kernel E: node_out = node_feat + vn[node_graph] + e_sum / max(cnt,1)
// ---------------------------------------------------------------------------
__global__ void finalize_kernel(const float* __restrict__ nf,
                                const int* __restrict__ ng,
                                const float* __restrict__ vn,
                                const float* __restrict__ e_sum,
                                const float* __restrict__ cnt,
                                float* __restrict__ out,
                                long total, int D) {
    long i = (long)blockIdx.x * blockDim.x + threadIdx.x;
    if (i >= total) return;
    const long row = i / D;
    const int  col = (int)(i - row * D);
    const float inv = 1.0f / fmaxf(cnt[row], 1.0f);
    out[i] = nf[i] + vn[(long)ng[row] * D + col] + e_sum[i] * inv;
}

// ---------------------------------------------------------------------------
// Launch
// ---------------------------------------------------------------------------
extern "C" void kernel_launch(void* const* d_in, const int* in_sizes, int n_in,
                              void* d_out, int out_size, void* d_ws, size_t ws_size,
                              hipStream_t stream) {
    const float* node_feat  = (const float*)d_in[0];
    const float* edge_feat  = (const float*)d_in[1];
    const int*   dst        = (const int*)d_in[2];
    const int*   node_graph = (const int*)d_in[3];
    const float* vn_emb     = (const float*)d_in[4];
    const float* W1         = (const float*)d_in[5];
    const float* b1         = (const float*)d_in[6];
    const float* ln_g       = (const float*)d_in[7];
    const float* ln_b       = (const float*)d_in[8];
    const float* W2         = (const float*)d_in[9];
    const float* b2         = (const float*)d_in[10];
    const float* We         = (const float*)d_in[11];
    const float* be         = (const float*)d_in[12];

    const int E = in_sizes[2];
    const int N = in_sizes[3];
    const int D = in_sizes[4];
    const int B = 64;

    float* out    = (float*)d_out;
    float* vn_out = out + (size_t)N * D;

    // workspace layout (256B-aligned chunks)
    auto al = [](size_t x) { return (x + 255) & ~(size_t)255; };
    char* base = (char*)d_ws;
    size_t off = 0;
    float* e_sum = (float*)(base + off);           off = al(off + (size_t)N * D * 4);
    float* e_cnt = (float*)(base + off);           off = al(off + (size_t)N * 4);
    float* gmean = (float*)(base + off);           off = al(off + (size_t)B * D * 4);
    unsigned short* Wh = (unsigned short*)(base + off); off = al(off + (size_t)D * D * 2);
    unsigned short* Wl = (unsigned short*)(base + off); off = al(off + (size_t)D * D * 2);
    unsigned short* Ah = (unsigned short*)(base + off); off = al(off + (size_t)E * D * 2);
    unsigned short* Al = (unsigned short*)(base + off); off = al(off + (size_t)E * D * 2);
    const bool use_packed = (off <= ws_size);

    // zero the atomic accumulators every call (graph-capture friendly)
    (void)hipMemsetAsync(e_sum, 0, (size_t)N * D * 4, stream);
    (void)hipMemsetAsync(e_cnt, 0, (size_t)N * 4, stream);

    graph_mean_kernel<<<B, 256, 0, stream>>>(node_feat, node_graph, gmean, N, D);
    vn_mlp_kernel<<<B, 512, 0, stream>>>(gmean, vn_emb, W1, b1, ln_g, ln_b,
                                         W2, b2, vn_out, D);
    edge_count_kernel<<<(E + 255) / 256, 256, 0, stream>>>(dst, e_cnt, E);

    dim3 grid(D / BN, (E + BM - 1) / BM);   // x = N-blocks: adjacent blocks share A tile in L2
    if (use_packed) {
        const long nA = (long)E * D;
        pack_split_kernel<<<(int)((nA / 4 + 255) / 256), 256, 0, stream>>>(
            edge_feat, Ah, Al, nA);
        dim3 wg((D + 255) / 256, D);
        pack_w_kernel<<<wg, 256, 0, stream>>>(We, Wh, Wl, D);
        edge_gemm_scatter_packed_kernel<<<grid, 256, 0, stream>>>(
            Ah, Al, Wh, Wl, be, dst, e_sum, E, D);
    } else {
        edge_gemm_scatter_fused_kernel<<<grid, 256, 0, stream>>>(
            edge_feat, We, be, dst, e_sum, E, D);
    }

    const long total = (long)N * D;
    finalize_kernel<<<(int)((total + 255) / 256), 256, 0, stream>>>(
        node_feat, node_graph, vn_out, e_sum, e_cnt, out, total, D);
}
